// NER_BiLSTMNet_6330781794375
// MI455X (gfx1250) — matrix operand order repaired and checked by reading, not compiled, not run on hardware
//
#include <hip/hip_runtime.h>

// Problem constants (from reference): B=16384, T=20, D=256, H=256, 4H=1024
#define BB   16384
#define TT   20
#define DD   256
#define HH   256
#define G4H  1024

typedef __bf16 v16bf __attribute__((ext_vector_type(16)));
typedef __bf16 v8bf  __attribute__((ext_vector_type(8)));
typedef __bf16 v4bf  __attribute__((ext_vector_type(4)));
typedef float  v8f   __attribute__((ext_vector_type(8)));
typedef float  v4f   __attribute__((ext_vector_type(4)));

// Tensor Data Mover path (gfx1250). Falls back to manual LDS staging when the
// builtin is unavailable (e.g. host pass). Arity: 5 args on ROCm7.2/clang-22,
// 6 args on amdgpu-toolchain/clang-23 (per probe doc).
#if defined(__has_builtin)
#if __has_builtin(__builtin_amdgcn_tensor_load_to_lds) && \
    __has_builtin(__builtin_amdgcn_s_wait_tensorcnt)
#define USE_TDM 1
#endif
#endif
#ifndef USE_TDM
#define USE_TDM 0
#endif

typedef unsigned int u32x4 __attribute__((ext_vector_type(4)));
typedef int          i32x4 __attribute__((ext_vector_type(4)));
typedef int          i32x8 __attribute__((ext_vector_type(8)));

static __device__ __forceinline__ __bf16 f2bf(float f) {
    unsigned u = __builtin_bit_cast(unsigned, f);
    u += 0x7FFFu + ((u >> 16) & 1u);          // round-to-nearest-even
    unsigned short s = (unsigned short)(u >> 16);
    return __builtin_bit_cast(__bf16, s);
}

static __device__ __forceinline__ float sigf(float x) {
    return 1.0f / (1.0f + __expf(-x));
}
static __device__ __forceinline__ float tanhfast(float x) {
    return 1.0f - 2.0f / (__expf(2.0f * x) + 1.0f);
}

// ---------------------------------------------------------------------------
// Utility kernels
// ---------------------------------------------------------------------------
__global__ void cvt_f32_bf16_kernel(const float* __restrict__ src,
                                    __bf16* __restrict__ dst, int n) {
    int i = blockIdx.x * blockDim.x + threadIdx.x;
    if (i < n) dst[i] = f2bf(src[i]);
}

__global__ void bias_sum_kernel(const float* __restrict__ a,
                                const float* __restrict__ b,
                                float* __restrict__ o, int n) {
    int i = blockIdx.x * blockDim.x + threadIdx.x;
    if (i < n) o[i] = a[i] + b[i];
}

__global__ void zero_u32_kernel(unsigned* __restrict__ p, int n) {
    int i = blockIdx.x * blockDim.x + threadIdx.x;
    if (i < n) p[i] = 0u;
}

// x [B,T,D] fp32  ->  xb [T*B, D] bf16  (transpose + convert, once)
__global__ __launch_bounds__(256)
void xpose_cvt_kernel(const float* __restrict__ x, __bf16* __restrict__ xb) {
    const long long i = (long long)blockIdx.x * blockDim.x + threadIdx.x; // BT*D/4
    const int k4 = (int)(i & 63);            // 64 quads per 256-wide row
    const long long m = i >> 6;              // m = t*B + b
    const int t = (int)(m >> 14);
    const int b = (int)(m & 16383);
    v4f v = *(const v4f*)(x + ((size_t)b * TT + t) * DD + k4 * 4);
    v4bf o;
#pragma unroll
    for (int j = 0; j < 4; ++j) o[j] = f2bf(v[j]);
    *(v4bf*)(xb + (size_t)m * DD + k4 * 4) = o;
}

// ---------------------------------------------------------------------------
// Unified WMMA GEMM:  Gout[M,1024] = A[M,256](bf16) @ W[1024,256]^T (+ Cin)
//   Block: 256 thr / 8 waves, tile 128(M) x 128(N), full K=256.
//   B panel (128 x 256 bf16, rows padded to 264) staged in LDS via TDM
//   (tensor_load_to_lds, pad 4 dwords every 128 dwords) or manual fallback.
//   Wave: 16x128 strip = 8 accumulator tiles, v_wmma_f32_16x16x32_bf16.
//   Per K-chunk: batch all 16 ds_loads, then 8 back-to-back WMMAs.
// ---------------------------------------------------------------------------
__global__ __launch_bounds__(256)
void wmma_gemm_kernel(const __bf16* __restrict__ A,    // [M,256] bf16
                      const __bf16* __restrict__ W,    // [1024,256] bf16
                      const float* __restrict__ Cin,   // [M,1024] or nullptr
                      float* __restrict__ Gout) {      // [M,1024]
    __shared__ __bf16 Bs[128 * 264];                   // padded rows: 132 dwords

    const int lane = threadIdx.x & 31;
    const int wv   = threadIdx.x >> 5;
    const int hi   = lane >> 4;               // lane half (K-group select)
    const int lm   = lane & 15;

    const int m0 = blockIdx.x * 128 + wv * 16;
    const int n0 = blockIdx.y * 128;

    // ---- stage B panel into LDS ----
#if USE_TDM
    // Statically-present but dynamically-dead store: keeps the compiler from
    // folding Bs loads to poison (the only real writer is the opaque TDM op).
    if ((size_t)Gout == 1) Bs[threadIdx.x] = (__bf16)1.0f;
    if (threadIdx.x < 32) {                   // wave 0 issues the DMA
        const unsigned long long ga =
            (unsigned long long)(const void*)(W + (size_t)n0 * DD);
        // D# group 0: count=1, lds_addr=0 (Bs is at LDS base), 57b global
        // address in bits [120:64], type=2 ("image") in [127:126].
        u32x4 g0 = { 1u,
                     0u,
                     (unsigned)(ga & 0xFFFFFFFFu),
                     (unsigned)((ga >> 32) & 0x01FFFFFFu) | (2u << 30) };
        // D# group 1: data_size=1(2B), pad_enable, pad_interval=6 (128 dw),
        // pad_amount=3 (4 dw) -> LDS row stride 264 bf16. tensor 256x1024,
        // tile 256x128, dim0 stride 256 elements.
        i32x8 g1 = { (int)((1u << 16) | (1u << 20) | (6u << 22) | (3u << 25)),
                     (int)(256u << 16),       // tensor_dim0 = 256
                     (int)(1024u << 16),      // tensor_dim1 = 1024
                     (int)(256u << 16),       // tile_dim0  = 256
                     128,                     // tile_dim1  = 128
                     256,                     // tensor_dim0_stride = 256
                     0, 0 };
        i32x4 gz = { 0, 0, 0, 0 };
#if __clang_major__ >= 23
        i32x8 gz8 = { 0, 0, 0, 0, 0, 0, 0, 0 };
        __builtin_amdgcn_tensor_load_to_lds(g0, g1, gz, gz, gz8, 0);
#else
        __builtin_amdgcn_tensor_load_to_lds(g0, g1, gz, gz, 0);
#endif
        __builtin_amdgcn_s_wait_tensorcnt(0);
    }
    __syncthreads();
#else
    {
        const int n  = threadIdx.x >> 1;              // 128 rows
        const int ks = (threadIdx.x & 1) * 128;       // 2 segments of 128 bf16
        const __bf16* src = W + (size_t)(n0 + n) * DD + ks;
        __bf16* dst = Bs + n * 264 + ks;
#pragma unroll
        for (int q = 0; q < 16; ++q)
            *(v8bf*)(dst + q * 8) = *(const v8bf*)(src + q * 8);
    }
    __syncthreads();
#endif

    const __bf16* arow = A + (size_t)(m0 + lm) * DD;

    v8f acc[8];
    if (Cin) {
#pragma unroll
        for (int nt = 0; nt < 8; ++nt) {
            const int col = n0 + nt * 16 + lm;
#pragma unroll
            for (int v = 0; v < 8; ++v)
                acc[nt][v] = Cin[(size_t)(m0 + v + 8 * hi) * G4H + col];
        }
    } else {
#pragma unroll
        for (int nt = 0; nt < 8; ++nt)
#pragma unroll
            for (int v = 0; v < 8; ++v) acc[nt][v] = 0.0f;
    }

#pragma unroll
    for (int k0 = 0; k0 < DD; k0 += 32) {
        const int kb = hi * 8;
        // A fragment: lanes 0-15 K={kb..kb+7, 16+kb..}, per ISA 16-bit A layout
        v8bf a0 = *(const v8bf*)(arow + k0 + kb);
        v8bf a1 = *(const v8bf*)(arow + k0 + 16 + kb);
        v16bf Av;
#pragma unroll
        for (int i = 0; i < 8; ++i) { Av[i] = a0[i]; Av[8 + i] = a1[i]; }

        // Preload ALL B fragments for this K-chunk (16 ds_loads in one burst),
        // then issue the 8 independent WMMAs back-to-back.
        v16bf Bv[8];
#pragma unroll
        for (int nt = 0; nt < 8; ++nt) {
            const __bf16* bp = Bs + (nt * 16 + lm) * 264 + k0 + hi * 8;
            v8bf b0 = *(const v8bf*)(bp);
            v8bf b1 = *(const v8bf*)(bp + 16);
#pragma unroll
            for (int i = 0; i < 8; ++i) { Bv[nt][i] = b0[i]; Bv[nt][8 + i] = b1[i]; }
        }
#pragma unroll
        for (int nt = 0; nt < 8; ++nt) {
            acc[nt] = __builtin_amdgcn_wmma_f32_16x16x32_bf16(
                false, Av, false, Bv[nt], (short)0, acc[nt], false, false);
        }
    }

#pragma unroll
    for (int nt = 0; nt < 8; ++nt) {
        const int col = n0 + nt * 16 + lm;
#pragma unroll
        for (int v = 0; v < 8; ++v)
            Gout[(size_t)(m0 + v + 8 * hi) * G4H + col] = acc[nt][v];
    }
}

// ---------------------------------------------------------------------------
// Gates + state update, 4 hidden units per thread.
// ---------------------------------------------------------------------------
__global__ __launch_bounds__(256)
void lstm_gate_kernel(const float* __restrict__ G,      // [B,1024]
                      const float* __restrict__ bias,   // [1024] b_ih+b_hh
                      float* __restrict__ c,            // [B,256] state
                      __bf16* __restrict__ h_bf,        // [B,256] next A operand
                      float* __restrict__ hs_out) {     // [B,256] hs slab
    const int idx = blockIdx.x * blockDim.x + threadIdx.x;   // B*H/4
    const int b = idx >> 6;
    const int q = (idx & 63) * 4;
    const float* Gr = G + (size_t)b * G4H;

    v4f gi = *(const v4f*)(Gr + q);
    v4f gf = *(const v4f*)(Gr + 256 + q);
    v4f gg = *(const v4f*)(Gr + 512 + q);
    v4f go = *(const v4f*)(Gr + 768 + q);
    v4f bi = *(const v4f*)(bias + q);
    v4f bf = *(const v4f*)(bias + 256 + q);
    v4f bg = *(const v4f*)(bias + 512 + q);
    v4f bo = *(const v4f*)(bias + 768 + q);
    v4f cp = *(const v4f*)(c + (size_t)b * HH + q);

    v4f cn, hn;
    v4bf hb;
#pragma unroll
    for (int i = 0; i < 4; ++i) {
        const float ig = sigf(gi[i] + bi[i]);
        const float fg = sigf(gf[i] + bf[i]);
        const float g2 = tanhfast(gg[i] + bg[i]);
        const float og = sigf(go[i] + bo[i]);
        const float cc = fg * cp[i] + ig * g2;
        cn[i] = cc;
        const float hh = og * tanhfast(cc);
        hn[i] = hh;
        hb[i] = f2bf(hh);
    }
    *(v4f*)(c + (size_t)b * HH + q)       = cn;
    *(v4bf*)(h_bf + (size_t)b * HH + q)   = hb;
    *(v4f*)(hs_out + (size_t)b * HH + q)  = hn;
}

// ---------------------------------------------------------------------------
// Softmax over 512 features of [hs_f[t,b,:], hs_b[t,b,:]] -> out[B,T,512].
// One wave per (t,b) row, 16 values per lane, shfl-xor reductions.
// ---------------------------------------------------------------------------
__global__ __launch_bounds__(256)
void softmax_kernel(const float* __restrict__ hs_f,     // [T,B,256]
                    const float* __restrict__ hs_b,     // [T,B,256]
                    float* __restrict__ out) {          // [B,T,512]
    const int lane = threadIdx.x & 31;
    const int wv   = threadIdx.x >> 5;
    const long long row = (long long)blockIdx.x * 8 + wv;   // row = t*B + b
    const int t = (int)(row >> 14);
    const int b = (int)(row & 16383);

    const float* base = (lane < 16)
        ? hs_f + (size_t)row * HH + lane * 16
        : hs_b + (size_t)row * HH + (lane - 16) * 16;

    float v[16];
    v4f* vv = (v4f*)v;
#pragma unroll
    for (int i = 0; i < 4; ++i) vv[i] = *(const v4f*)(base + i * 4);

    float mx = v[0];
#pragma unroll
    for (int i = 1; i < 16; ++i) mx = fmaxf(mx, v[i]);
#pragma unroll
    for (int off = 16; off >= 1; off >>= 1)
        mx = fmaxf(mx, __shfl_xor(mx, off, 32));

    float sum = 0.0f;
#pragma unroll
    for (int i = 0; i < 16; ++i) { v[i] = __expf(v[i] - mx); sum += v[i]; }
#pragma unroll
    for (int off = 16; off >= 1; off >>= 1)
        sum += __shfl_xor(sum, off, 32);
    const float inv = 1.0f / sum;
#pragma unroll
    for (int i = 0; i < 16; ++i) v[i] *= inv;

    float* o = out + ((size_t)b * TT + t) * 512 + lane * 16;
#pragma unroll
    for (int i = 0; i < 4; ++i) *(v4f*)(o + i * 4) = vv[i];
}

// ---------------------------------------------------------------------------
// Host launcher
// ---------------------------------------------------------------------------
extern "C" void kernel_launch(void* const* d_in, const int* in_sizes, int n_in,
                              void* d_out, int out_size, void* d_ws, size_t ws_size,
                              hipStream_t stream) {
    const float* x      = (const float*)d_in[0];   // [B,T,D]
    const float* W_ih_f = (const float*)d_in[1];   // [1024,256]
    const float* W_hh_f = (const float*)d_in[2];   // [1024,256]
    const float* b_ih_f = (const float*)d_in[3];
    const float* b_hh_f = (const float*)d_in[4];
    const float* W_ih_b = (const float*)d_in[5];
    const float* W_hh_b = (const float*)d_in[6];
    const float* b_ih_b = (const float*)d_in[7];
    const float* b_hh_b = (const float*)d_in[8];
    float* out = (float*)d_out;                    // [B,T,512]

    // ---- carve workspace ----
    char* p = (char*)d_ws;
    auto take = [&](size_t bytes) -> void* {
        void* r = (void*)p;
        p += (bytes + 255) & ~(size_t)255;
        return r;
    };
    const size_t WSZ = (size_t)G4H * DD;            // 262144 elems / weight
    __bf16* Wihf_bf = (__bf16*)take(WSZ * 2);
    __bf16* Whhf_bf = (__bf16*)take(WSZ * 2);
    __bf16* Wihb_bf = (__bf16*)take(WSZ * 2);
    __bf16* Whhb_bf = (__bf16*)take(WSZ * 2);
    float*  bias_f  = (float*)take(G4H * 4);
    float*  bias_b  = (float*)take(G4H * 4);
    float*  c_f     = (float*)take((size_t)BB * HH * 4);
    float*  c_b     = (float*)take((size_t)BB * HH * 4);
    __bf16* h_f_bf  = (__bf16*)take((size_t)BB * HH * 2);
    __bf16* h_b_bf  = (__bf16*)take((size_t)BB * HH * 2);
    __bf16* x_bf    = (__bf16*)take((size_t)TT * BB * DD * 2);   // [T*B, D]
    float*  G_f     = (float*)take((size_t)BB * G4H * 4);
    float*  G_b     = (float*)take((size_t)BB * G4H * 4);
    float*  hs_f    = (float*)take((size_t)TT * BB * HH * 4);
    float*  hs_b    = (float*)take((size_t)TT * BB * HH * 4);
    float*  Gx_f    = (float*)take((size_t)TT * BB * G4H * 4);
    float*  Gx_b    = (float*)take((size_t)TT * BB * G4H * 4);

    // ---- prep: weight conversion + bias sums + zero state + x transpose ----
    cvt_f32_bf16_kernel<<<(int)(WSZ / 256), 256, 0, stream>>>(W_ih_f, Wihf_bf, (int)WSZ);
    cvt_f32_bf16_kernel<<<(int)(WSZ / 256), 256, 0, stream>>>(W_hh_f, Whhf_bf, (int)WSZ);
    cvt_f32_bf16_kernel<<<(int)(WSZ / 256), 256, 0, stream>>>(W_ih_b, Wihb_bf, (int)WSZ);
    cvt_f32_bf16_kernel<<<(int)(WSZ / 256), 256, 0, stream>>>(W_hh_b, Whhb_bf, (int)WSZ);
    bias_sum_kernel<<<4, 256, 0, stream>>>(b_ih_f, b_hh_f, bias_f, G4H);
    bias_sum_kernel<<<4, 256, 0, stream>>>(b_ih_b, b_hh_b, bias_b, G4H);

    const int stateWords = BB * HH;          // f32 words
    const int hbfWords   = BB * HH / 2;      // bf16 buffer in u32 words
    zero_u32_kernel<<<stateWords / 256, 256, 0, stream>>>((unsigned*)c_f, stateWords);
    zero_u32_kernel<<<stateWords / 256, 256, 0, stream>>>((unsigned*)c_b, stateWords);
    zero_u32_kernel<<<hbfWords / 256, 256, 0, stream>>>((unsigned*)h_f_bf, hbfWords);
    zero_u32_kernel<<<hbfWords / 256, 256, 0, stream>>>((unsigned*)h_b_bf, hbfWords);

    xpose_cvt_kernel<<<(BB * TT * DD / 4) / 256, 256, 0, stream>>>(x, x_bf);

    // ---- phase 1: input-projection GEMMs (no sequential dependency) ----
    {
        dim3 gih((BB * TT) / 128, G4H / 128);
        wmma_gemm_kernel<<<gih, 256, 0, stream>>>(x_bf, Wihf_bf, nullptr, Gx_f);
        wmma_gemm_kernel<<<gih, 256, 0, stream>>>(x_bf, Wihb_bf, nullptr, Gx_b);
    }

    // ---- phase 2: recurrence (backward consumes Gx_b time-reversed,
    //      stores hs_b in scan order -> matches reference concat) ----
    dim3 ghh(BB / 128, G4H / 128);
    const int gateBlocks = (BB * HH / 4) / 256;
    for (int t = 0; t < TT; ++t) {
        wmma_gemm_kernel<<<ghh, 256, 0, stream>>>(
            h_f_bf, Whhf_bf, Gx_f + (size_t)t * BB * G4H, G_f);
        lstm_gate_kernel<<<gateBlocks, 256, 0, stream>>>(
            G_f, bias_f, c_f, h_f_bf, hs_f + (size_t)t * BB * HH);
        wmma_gemm_kernel<<<ghh, 256, 0, stream>>>(
            h_b_bf, Whhb_bf, Gx_b + (size_t)(TT - 1 - t) * BB * G4H, G_b);
        lstm_gate_kernel<<<gateBlocks, 256, 0, stream>>>(
            G_b, bias_b, c_b, h_b_bf, hs_b + (size_t)t * BB * HH);
    }

    // ---- phase 3: softmax over concatenated features -> [B,T,512] ----
    softmax_kernel<<<(BB * TT) / 8, 256, 0, stream>>>(hs_f, hs_b, out);

    (void)in_sizes; (void)n_in; (void)out_size; (void)ws_size;
}
